// CRF_32779190403443
// MI455X (gfx1250) — matrix operand order repaired and checked by reading, not compiled, not run on hardware
//
#include <hip/hip_runtime.h>
#include <hip/hip_bf16.h>

// CRF log-likelihood for MI455X (gfx1250, wave32).
// Forward scan as per-step GEMM in base-2 log domain:
//   alpha2' = m2 + log2( P @ expT ) + em*log2e,  P = exp2(alpha2 - m2)
// V_WMMA_F32_16X16X32_F16, K=64 via 2 chained WMMAs. Raw v_exp_f32/v_log_f32.
// P computed once per block (8 exps/thread) and shared through LDS in the
// exact WMMA A-fragment layout; 2 barriers per step.

typedef __attribute__((ext_vector_type(16))) _Float16 v16h;
typedef __attribute__((ext_vector_type(8)))  _Float16 v8h;
typedef __attribute__((ext_vector_type(8)))  float    v8f;

#define S_LEN 512
#define B_SZ  512
#define T_TAG 64
#define LOG2E 1.44269504088896340736f
#define LN2   0.69314718055994530942f
#define ASTR  68   // alpha row stride (floats): 16B-aligned rows, conflict-free
#define PSTR  72   // p row stride (halves) = 36 dwords: 16B-aligned, conflict-free

// ---------------------------------------------------------------------------
// Forward algorithm (denominator). grid = B/16 = 32 blocks, 128 threads.
// Block owns batch rows [b0,b0+16); wave w owns tag columns [16w,16w+16).
// ---------------------------------------------------------------------------
__global__ __launch_bounds__(128) void crf_forward_kernel(
    const float*         __restrict__ emissions,   // [S, B, T]
    const unsigned char* __restrict__ mask,        // [S, B] (bool)
    const float*         __restrict__ startT,      // [T]
    const float*         __restrict__ endT,        // [T]
    const float*         __restrict__ trans,       // [T, T]
    float*               __restrict__ den)         // [B]
{
    __shared__ __align__(16) float    s_alpha[16][ASTR];
    __shared__ __align__(16) _Float16 s_p[16][PSTR];
    __shared__ __align__(16) float    s_m[16];
    __shared__ float s_end2[64];

    const int tid  = threadIdx.x;
    const int wave = tid >> 5;
    const int lane = tid & 31;
    const int r    = lane & 15;     // A/B fragment row
    const int hi   = lane >> 4;     // lane half
    const int n0   = wave * 16;     // N-block start
    const int b0   = blockIdx.x * 16;
    const int colc = n0 + r;        // C/D column
    const int prow = tid >> 3;      // producer row  (stage 1)
    const int pc0  = (tid & 7) * 8; // producer col base (8 cols)

    // ---- Constant B fragments: exp(T) = exp2(T*log2e), f16, columns [n0,n0+16).
    // B in contiguous k order (k = 16*hi + s): VGPR v: lanes 0-15 K=2v,2v+1; lanes 16-31 K=16+2v,+1.
    v16h Blo, Bhi;
    {
        const int khalf = hi * 16;
#pragma unroll
        for (int v = 0; v < 8; ++v) {
            const int k = khalf + 2 * v;
            Blo[2*v]   = (_Float16)__builtin_amdgcn_exp2f(trans[(k        )*T_TAG + colc] * LOG2E);
            Blo[2*v+1] = (_Float16)__builtin_amdgcn_exp2f(trans[(k + 1    )*T_TAG + colc] * LOG2E);
            Bhi[2*v]   = (_Float16)__builtin_amdgcn_exp2f(trans[(32 + k    )*T_TAG + colc] * LOG2E);
            Bhi[2*v+1] = (_Float16)__builtin_amdgcn_exp2f(trans[(32 + k + 1)*T_TAG + colc] * LOG2E);
        }
    }
    if (tid < 64) s_end2[tid] = endT[tid] * LOG2E;

    // ---- alpha2_0 = (em[0] + start) * log2e  (1024 contiguous f32, coalesced)
    {
        const float* e0 = emissions + (size_t)b0 * T_TAG;
#pragma unroll
        for (int i = 0; i < 8; ++i) {
            const int idx = tid * 8 + i;
            s_alpha[idx >> 6][idx & 63] = (e0[idx] + startT[idx & 63]) * LOG2E;
        }
    }
    __syncthreads();

    // Persistent copy of this lane's C-layout alpha values (rows v+8*hi, col colc).
    float alpha_reg[8];
#pragma unroll
    for (int v = 0; v < 8; ++v) alpha_reg[v] = s_alpha[v + 8*hi][colc];

    for (int t = 1; t < S_LEN; ++t) {
        const float* et = emissions + ((size_t)t * B_SZ + b0) * T_TAG;

        if (t + 1 < S_LEN && tid < 32) {
            __builtin_prefetch(emissions + ((size_t)(t + 1) * B_SZ + b0) * T_TAG + tid * 32, 0, 0);
        }

        // Hoist this step's global loads so they overlap stage 1.
        float emv[8];
#pragma unroll
        for (int v = 0; v < 8; ++v)
            emv[v] = et[(size_t)(v + 8*hi) * T_TAG + colc];
        const unsigned long long mk8 =
            *(const unsigned long long*)(mask + (size_t)t * B_SZ + b0 + 8 * hi);

        // ================= Stage 1: produce P once per block =================
        // Thread owns row prow, cols pc0..pc0+7.
        const float4 a0 = *(const float4*)(&s_alpha[prow][pc0]);
        const float4 a1 = *(const float4*)(&s_alpha[prow][pc0 + 4]);

        float pm = fmaxf(fmaxf(fmaxf(a0.x, a0.y), fmaxf(a0.z, a0.w)),
                         fmaxf(fmaxf(a1.x, a1.y), fmaxf(a1.z, a1.w)));
        pm = fmaxf(pm, __shfl_xor(pm, 1, 32));
        pm = fmaxf(pm, __shfl_xor(pm, 2, 32));
        pm = fmaxf(pm, __shfl_xor(pm, 4, 32));   // full row max in all 8 owners
        if ((tid & 7) == 0) s_m[prow] = pm;

        v8h pv;
        pv[0] = (_Float16)__builtin_amdgcn_exp2f(a0.x - pm);
        pv[1] = (_Float16)__builtin_amdgcn_exp2f(a0.y - pm);
        pv[2] = (_Float16)__builtin_amdgcn_exp2f(a0.z - pm);
        pv[3] = (_Float16)__builtin_amdgcn_exp2f(a0.w - pm);
        pv[4] = (_Float16)__builtin_amdgcn_exp2f(a1.x - pm);
        pv[5] = (_Float16)__builtin_amdgcn_exp2f(a1.y - pm);
        pv[6] = (_Float16)__builtin_amdgcn_exp2f(a1.z - pm);
        pv[7] = (_Float16)__builtin_amdgcn_exp2f(a1.w - pm);
        *(v8h*)(&s_p[prow][pc0]) = pv;

        __syncthreads();   // P, s_m visible; all alpha reads of this step done

        // ================= Stage 2: WMMA + log-space update ==================
        // A frags: contiguous k order (k = 16*hi + s), as for B -> contiguous 8-half runs in s_p.
        const v8h plo0 = *(const v8h*)(&s_p[r][16 * hi]);       // K =      16hi..+7
        const v8h plo1 = *(const v8h*)(&s_p[r][16 * hi + 8]);   // K =  8 + 16hi..+7
        const v8h phi0 = *(const v8h*)(&s_p[r][32 + 16 * hi]);  // K = 32 + 16hi..+7
        const v8h phi1 = *(const v8h*)(&s_p[r][40 + 16 * hi]);  // K = 40 + 16hi..+7
        const v16h Alo = __builtin_shufflevector(plo0, plo1,
            0,1,2,3,4,5,6,7, 8,9,10,11,12,13,14,15);
        const v16h Ahi = __builtin_shufflevector(phi0, phi1,
            0,1,2,3,4,5,6,7, 8,9,10,11,12,13,14,15);

        v8f acc = {};
        acc = __builtin_amdgcn_wmma_f32_16x16x32_f16(false, Alo, false, Blo,
                                                     (short)0, acc, false, false);
        acc = __builtin_amdgcn_wmma_f32_16x16x32_f16(false, Ahi, false, Bhi,
                                                     (short)0, acc, false, false);

        // Row maxima for this lane's C rows (8hi..8hi+7): two float4 loads.
        const float4 ms0 = *(const float4*)(&s_m[8 * hi]);
        const float4 ms1 = *(const float4*)(&s_m[8 * hi + 4]);
        const float mrow[8] = { ms0.x, ms0.y, ms0.z, ms0.w,
                                ms1.x, ms1.y, ms1.z, ms1.w };

        // alpha2' = m[row] + log2(acc) + em*log2e   (C/D: row=v+8*hi, col=colc)
        float nv[8];
#pragma unroll
        for (int v = 0; v < 8; ++v) {
            const float upd = fmaf(emv[v], LOG2E,
                                   mrow[v] + __builtin_amdgcn_logf(acc[v]));
            nv[v] = ((mk8 >> (8 * v)) & 0xffull) ? upd : alpha_reg[v];
        }
#pragma unroll
        for (int v = 0; v < 8; ++v) {
            s_alpha[v + 8*hi][colc] = nv[v];
            alpha_reg[v] = nv[v];
        }
        __syncthreads();   // alpha' visible for next step's stage 1
    }

    // ---- denominator[b] = ln2 * logsumexp2_j(alpha2[b,j] + end2[j])
    if (tid < 16) {
        float mx = -3.4e38f;
        for (int j = 0; j < 64; ++j) mx = fmaxf(mx, s_alpha[tid][j] + s_end2[j]);
        float s = 0.f;
        for (int j = 0; j < 64; ++j)
            s += __builtin_amdgcn_exp2f(s_alpha[tid][j] + s_end2[j] - mx);
        den[b0 + tid] = (mx + __builtin_amdgcn_logf(s)) * LN2;
    }
}

// ---------------------------------------------------------------------------
// Numerator: one thread per batch element (gather-based path score).
// ---------------------------------------------------------------------------
__global__ __launch_bounds__(256) void crf_numerator_kernel(
    const float*         __restrict__ emissions,
    const int*           __restrict__ tags,      // [S, B]
    const unsigned char* __restrict__ mask,      // [S, B]
    const float*         __restrict__ startT,
    const float*         __restrict__ endT,
    const float*         __restrict__ trans,
    float*               __restrict__ num)
{
    const int b = blockIdx.x * blockDim.x + threadIdx.x;
    if (b >= B_SZ) return;

    int t0 = tags[b];
    float score = startT[t0] + emissions[(size_t)b * T_TAG + t0];
    int prev = t0;
    int msum = (mask[b] != 0) ? 1 : 0;

    for (int t = 1; t < S_LEN; ++t) {
        const int   cur = tags[t * B_SZ + b];
        const float mf  = (mask[t * B_SZ + b] != 0) ? 1.0f : 0.0f;
        score += (trans[prev * T_TAG + cur] +
                  emissions[((size_t)t * B_SZ + b) * T_TAG + cur]) * mf;
        msum += (mask[t * B_SZ + b] != 0) ? 1 : 0;
        prev = cur;
    }
    const int last = tags[(msum - 1) * B_SZ + b];
    num[b] = score + endT[last];
}

// ---------------------------------------------------------------------------
// Final reduction: mean(numerator - denominator)
// ---------------------------------------------------------------------------
__global__ __launch_bounds__(256) void crf_reduce_kernel(
    const float* __restrict__ num,
    const float* __restrict__ den,
    float*       __restrict__ out)
{
    __shared__ float s[256];
    const int tid = threadIdx.x;
    float v = (num[tid] - den[tid]) + (num[tid + 256] - den[tid + 256]);
    s[tid] = v;
    __syncthreads();
#pragma unroll
    for (int off = 128; off > 0; off >>= 1) {
        if (tid < off) s[tid] += s[tid + off];
        __syncthreads();
    }
    if (tid == 0) out[0] = s[0] / (float)B_SZ;
}

// ---------------------------------------------------------------------------
extern "C" void kernel_launch(void* const* d_in, const int* in_sizes, int n_in,
                              void* d_out, int out_size, void* d_ws, size_t ws_size,
                              hipStream_t stream) {
    const float*         emissions = (const float*)d_in[0];
    const int*           tags      = (const int*)d_in[1];           // integer input
    const unsigned char* mask      = (const unsigned char*)d_in[2]; // bool, 1 byte/elem
    const float*         startT    = (const float*)d_in[3];
    const float*         endT      = (const float*)d_in[4];
    const float*         trans     = (const float*)d_in[5];
    float*               out       = (float*)d_out;

    float* den = (float*)d_ws;          // [B]
    float* num = den + B_SZ;            // [B]

    crf_forward_kernel<<<dim3(B_SZ / 16), dim3(128), 0, stream>>>(
        emissions, mask, startT, endT, trans, den);
    crf_numerator_kernel<<<dim3(B_SZ / 256), dim3(256), 0, stream>>>(
        emissions, tags, mask, startT, endT, trans, num);
    crf_reduce_kernel<<<dim3(1), dim3(256), 0, stream>>>(num, den, out);

    (void)in_sizes; (void)n_in; (void)out_size; (void)ws_size;
}
